// Colorizer_53893249630373
// MI455X (gfx1250) — matrix operand order repaired and checked by reading, not compiled, not run on hardware
//
#include <hip/hip_runtime.h>
#include <hip/hip_bf16.h>

// ---------------------------------------------------------------------------
// MI455X (gfx1250) implementation of the SIFT->CNN->bilateral colorizer.
// fp16 NHWC activations; all large convs as implicit-GEMM on
// V_WMMA_F32_16X16X32_F16 (wave32), fp32 accumulate; scalar tail kernels.
// Halo handling via a zeroed scratch page (branchless A loads); each wave
// computes two 16-pixel M tiles sharing one B operand. Per tap, all KC
// chunk loads (A0/A1/B) are hoisted ahead of the WMMA chain so the
// scheduler can overlap load latency with matrix ops (staggered
// s_wait_loadcnt instead of draining to zero per WMMA).
// ---------------------------------------------------------------------------

typedef __attribute__((ext_vector_type(16))) _Float16 v16h;
typedef __attribute__((ext_vector_type(8)))  _Float16 v8h;
typedef __attribute__((ext_vector_type(8)))  float    v8f;

#define PI_F 3.14159265358979323846f

// ---------------------------------------------------------------------------
// Small helpers
// ---------------------------------------------------------------------------
__device__ inline float wave_red_add(float v) {
#pragma unroll
    for (int m = 16; m >= 1; m >>= 1) v += __shfl_xor(v, m, 32);
    return v;
}

__device__ inline int reflect_idx(int i, int n) {   // jnp.pad 'reflect' (no edge repeat)
    if (i < 0) i = -i;
    if (i >= n) i = 2 * n - 2 - i;
    return i;
}

// ---------------------------------------------------------------------------
// Init: zero groupnorm stats accumulators + the zero-page used for conv halos
// (harness poisons ws with 0xAA before timing, so this runs every launch)
// ---------------------------------------------------------------------------
__global__ void init_kernel(float* stats, float* zpage) {
    stats[threadIdx.x] = 0.f;
    if (threadIdx.x < 128) zpage[threadIdx.x] = 0.f;   // 512 B of zeros
}

// ---------------------------------------------------------------------------
// Dense SIFT stage 1: soft orientation histogram -> ang (B,8,H,W) f32 planar
// ---------------------------------------------------------------------------
__global__ void sift_ang_kernel(const float* __restrict__ x, float* __restrict__ ang,
                                int B, int H, int W) {
    size_t idx = (size_t)blockIdx.x * blockDim.x + threadIdx.x;
    size_t HW = (size_t)H * W;
    if (idx >= (size_t)B * HW) return;
    int b = (int)(idx / HW);
    int rem = (int)(idx % HW);
    int i = rem / W, j = rem % W;
    const float* xb = x + (size_t)b * HW;
    int jm = j > 0 ? j - 1 : 0, jp = j < W - 1 ? j + 1 : W - 1;
    int im = i > 0 ? i - 1 : 0, ip = i < H - 1 ? i + 1 : H - 1;
    float gx = (xb[(size_t)i * W + jp] - xb[(size_t)i * W + jm]) * 0.5f;
    float gy = (xb[(size_t)ip * W + j] - xb[(size_t)im * W + j]) * 0.5f;
    float mag = sqrtf(gx * gx + gy * gy + 1e-10f);
    float ori = atan2f(gy, gx + 1e-10f) + 2.f * PI_F;
    float ob = 8.f * ori / (2.f * PI_F);
    float bf = floorf(ob);
    float w1 = ob - bf;
    int b0 = ((int)bf) & 7;          // bf >= 0 so & is mod
    int b1 = (b0 + 1) & 7;
    float w0m = (1.f - w1) * mag, w1m = w1 * mag;
    float* ab = ang + (size_t)b * 8 * HW + (size_t)i * W + j;
#pragma unroll
    for (int c = 0; c < 8; ++c) {
        float v = (c == b0 ? w0m : 0.f) + (c == b1 ? w1m : 0.f);
        ab[(size_t)c * HW] = v;
    }
}

// ---------------------------------------------------------------------------
// Dense SIFT stage 2: depthwise 4x4 triangular pooling, pad 2 -> (B,8,H+1,W+1)
// ---------------------------------------------------------------------------
__global__ void sift_pool_kernel(const float* __restrict__ ang, float* __restrict__ pooled,
                                 int B, int H, int W) {
    int Hp = H + 1, Wp = W + 1;
    size_t total = (size_t)B * 8 * Hp * Wp;
    size_t idx = (size_t)blockIdx.x * blockDim.x + threadIdx.x;
    if (idx >= total) return;
    int j = (int)(idx % Wp);
    int i = (int)((idx / Wp) % Hp);
    int a = (int)((idx / ((size_t)Wp * Hp)) % 8);
    int b = (int)(idx / ((size_t)Wp * Hp * 8));
    const float pk[4] = {0.25f, 0.75f, 0.75f, 0.25f};   // (2-|t+0.5-2|)/2
    const float* ab = ang + ((size_t)b * 8 + a) * H * W;
    float acc = 0.f;
#pragma unroll
    for (int dy = 0; dy < 4; ++dy) {
        int iy = i + dy - 2;
        if ((unsigned)iy < (unsigned)H) {
#pragma unroll
            for (int dx = 0; dx < 4; ++dx) {
                int ix = j + dx - 2;
                if ((unsigned)ix < (unsigned)W)
                    acc += pk[dy] * pk[dx] * ab[(size_t)iy * W + ix];
            }
        }
    }
    pooled[idx] = acc;
}

// ---------------------------------------------------------------------------
// Dense SIFT stage 3: 4x4 patch gather + l2/clip/l2/rootsift normalize.
// One wave32 per pixel (4 channels per lane); writes s as NHWC fp16, C=128.
// ---------------------------------------------------------------------------
__global__ __launch_bounds__(128)
void sift_desc_kernel(const float* __restrict__ pooled, _Float16* __restrict__ s,
                      int B, int H, int W) {
    int lane = threadIdx.x & 31;
    int wv = threadIdx.x >> 5;
    size_t HW = (size_t)H * W;
    size_t pix = (size_t)blockIdx.x * 4 + wv;
    int b = (int)(pix / HW);
    int rem = (int)(pix % HW);
    int i = rem / W, j = rem % W;
    size_t HWp = (size_t)(H + 1) * (W + 1);
    float v[4];
#pragma unroll
    for (int t = 0; t < 4; ++t) {
        int c = lane * 4 + t;
        int a = c >> 4, r = c & 15, yy = r >> 2, xx = r & 3;
        int iy = i + yy - 1, ix = j + xx - 1;           // pooled dims (H+1, W+1)
        bool ok = (unsigned)iy <= (unsigned)H && (unsigned)ix <= (unsigned)W;
        v[t] = ok ? pooled[((size_t)b * 8 + a) * HWp + (size_t)iy * (W + 1) + ix] : 0.f;
    }
    float ssq = wave_red_add(v[0] * v[0] + v[1] * v[1] + v[2] * v[2] + v[3] * v[3]);
    float inv = 1.f / fmaxf(sqrtf(ssq), 1e-12f);
#pragma unroll
    for (int t = 0; t < 4; ++t) v[t] = fminf(fmaxf(v[t] * inv, 0.f), 0.2f);
    ssq = wave_red_add(v[0] * v[0] + v[1] * v[1] + v[2] * v[2] + v[3] * v[3]);
    inv = 1.f / fmaxf(sqrtf(ssq), 1e-12f);
#pragma unroll
    for (int t = 0; t < 4; ++t) v[t] *= inv;
    float l1 = wave_red_add(v[0] + v[1] + v[2] + v[3]);
    float il1 = 1.f / fmaxf(l1, 1e-12f);
    _Float16* sp = s + pix * 128 + lane * 4;
#pragma unroll
    for (int t = 0; t < 4; ++t) sp[t] = (_Float16)sqrtf(v[t] * il1 + 1e-10f);
}

// ---------------------------------------------------------------------------
// Weight packing: OIHW f32 -> fp16 WMMA B-operand layout
//   wp[tap][kc][nt][lane][16h] ; lane L holds column N = nt*16 + (L%16),
//   K = kc*32 + (L<16 ? 0 : 16) + h   (contiguous k order, same as the A loads)
// ---------------------------------------------------------------------------
__global__ void pack_weights_kernel(const float* __restrict__ w, _Float16* __restrict__ wp,
                                    int COUT, int CIN, int KS, int CINP, int NTILES) {
    int KC = CINP / 32;
    size_t total = (size_t)KS * KS * KC * NTILES * 32 * 16;
    size_t i = (size_t)blockIdx.x * blockDim.x + threadIdx.x;
    if (i >= total) return;
    int h = (int)(i % 16);
    int lane = (int)((i / 16) % 32);
    int nt = (int)((i / (16 * 32)) % NTILES);
    int kc = (int)((i / ((size_t)16 * 32 * NTILES)) % KC);
    int tap = (int)(i / ((size_t)16 * 32 * NTILES * KC));
    int ky = tap / KS, kx = tap % KS;
    int n = nt * 16 + (lane & 15);
    int k = kc * 32 + ((lane >> 4) * 16) + h;
    float val = (n < COUT && k < CIN) ? w[(((size_t)n * CIN + k) * KS + ky) * KS + kx] : 0.f;
    wp[i] = (_Float16)val;
}

// ---------------------------------------------------------------------------
// Implicit-GEMM conv via V_WMMA_F32_16X16X32_F16.
//   act: NHWC fp16, channel stride CINP (multiple of 32)
//   one wave = 2 x 16 consecutive pixels (same row) x 16 output channels;
//   B operand loaded once, used by both M tiles. Halo lanes read the zero
//   page (branchless: one 64-bit cndmask on the address per tap). All KC
//   chunk loads per tap are hoisted before the WMMA chain.
//   grid.x = B*H*(W/32), block = NTILES waves
// ---------------------------------------------------------------------------
template <int CINP, int KS, int PAD, int NTILES, int COUT>
__global__ __launch_bounds__(NTILES * 32)
void wmma_conv_kernel(const _Float16* __restrict__ act,
                      const _Float16* __restrict__ wp,
                      const float* __restrict__ bias,
                      _Float16* __restrict__ out,
                      const _Float16* __restrict__ zpage,
                      int B, int H, int W, int copOut) {
    const int KC = CINP / 32;
    const int WT2 = W >> 5;                 // 32-pixel supertiles per row
    int lane = threadIdx.x & 31;
    int nt = threadIdx.x >> 5;
    int mt = blockIdx.x;
    int x0 = (mt % WT2) << 5;
    int y = (mt / WT2) % H;
    int b = mt / (WT2 * H);
    int laneHi = lane >> 4;
    int mcol = lane & 15;
    int xa0 = x0 + mcol;                    // M tile 0 pixel column
    int xa1 = x0 + 16 + mcol;               // M tile 1 pixel column

    union A16 { v16h v; v8h h[2]; };
    v8f acc0 = {};
    v8f acc1 = {};

    for (int ky = 0; ky < KS; ++ky) {
        int iy = y + ky - PAD;
        bool rowok = (unsigned)iy < (unsigned)H;
        const _Float16* rowp = act + ((size_t)b * H + iy) * W * CINP;
        for (int kx = 0; kx < KS; ++kx) {
            int ix0 = xa0 + kx - PAD;
            int ix1 = xa1 + kx - PAD;
            bool ok0 = rowok && ((unsigned)ix0 < (unsigned)W);
            bool ok1 = rowok && ((unsigned)ix1 < (unsigned)W);
            const _Float16* ap0 = ok0 ? rowp + (size_t)ix0 * CINP : zpage;
            const _Float16* ap1 = ok1 ? rowp + (size_t)ix1 * CINP : zpage;
            const _Float16* wb = wp + ((size_t)(ky * KS + kx) * KC * NTILES) * 32 * 16;

            // Hoist all loads of this tap so the scheduler can issue them as
            // clauses and overlap the WMMA chain with outstanding loads.
            A16 a0[KC], a1[KC];
            v16h bm[KC];
#pragma unroll
            for (int kc = 0; kc < KC; ++kc) {
                int c0 = kc * 32 + laneHi * 16;
                a0[kc].h[0] = *(const v8h*)(ap0 + c0);        // K 0..7 / 16..23
                a0[kc].h[1] = *(const v8h*)(ap0 + c0 + 8);    // K 8..15 / 24..31
                a1[kc].h[0] = *(const v8h*)(ap1 + c0);
                a1[kc].h[1] = *(const v8h*)(ap1 + c0 + 8);
                bm[kc] = *(const v16h*)(wb + ((size_t)(kc * NTILES + nt) * 32 + lane) * 16);
            }
#pragma unroll
            for (int kc = 0; kc < KC; ++kc) {
                acc0 = __builtin_amdgcn_wmma_f32_16x16x32_f16(
                    false, a0[kc].v, false, bm[kc], (short)0, acc0, false, false);
                acc1 = __builtin_amdgcn_wmma_f32_16x16x32_f16(
                    false, a1[kc].v, false, bm[kc], (short)0, acc1, false, false);
            }
        }
    }
    int n = nt * 16 + mcol;
    float bv = (n < COUT) ? bias[n] : 0.f;
    size_t obase0 = (((size_t)b * H + y) * W + x0) * copOut + n;
    size_t obase1 = obase0 + (size_t)16 * copOut;
#pragma unroll
    for (int r = 0; r < 8; ++r) {
        int mm = r + laneHi * 8;                         // C/D layout: VGPR r -> M=r(+8)
        out[obase0 + (size_t)mm * copOut] = (_Float16)(acc0[r] + bv);
        out[obase1 + (size_t)mm * copOut] = (_Float16)(acc1[r] + bv);
    }
}

// ---------------------------------------------------------------------------
// GroupNorm statistics: per (b,g) sum/sumsq, LDS tree reduce + global atomics
// ---------------------------------------------------------------------------
__global__ __launch_bounds__(256)
void gn_stats_kernel(const _Float16* __restrict__ x, float* __restrict__ stats,
                     int HW, int CPAD, int C, int groups, int slices) {
    int bg = blockIdx.x / slices;
    int sl = blockIdx.x % slices;
    int b = bg / groups, g = bg % groups;
    int cg = C / groups;
    size_t total = (size_t)HW * cg;
    size_t chunk = (total + slices - 1) / slices;
    size_t beg = (size_t)sl * chunk;
    size_t end = beg + chunk; if (end > total) end = total;
    float s = 0.f, q = 0.f;
    const _Float16* xb = x + (size_t)b * HW * CPAD + (size_t)g * cg;
    for (size_t i = beg + threadIdx.x; i < end; i += blockDim.x) {
        size_t pix = i / cg;
        int c = (int)(i % cg);
        float v = (float)xb[pix * CPAD + c];
        s += v; q += v * v;
    }
    __shared__ float ss[256], sq[256];
    ss[threadIdx.x] = s; sq[threadIdx.x] = q;
    __syncthreads();
    for (int st = 128; st > 0; st >>= 1) {
        if ((int)threadIdx.x < st) {
            ss[threadIdx.x] += ss[threadIdx.x + st];
            sq[threadIdx.x] += sq[threadIdx.x + st];
        }
        __syncthreads();
    }
    if (threadIdx.x == 0) {
        atomicAdd(&stats[bg * 2 + 0], ss[0]);
        atomicAdd(&stats[bg * 2 + 1], sq[0]);
    }
}

// ---------------------------------------------------------------------------
// GroupNorm normalize (+optional ReLU), in place; zero-fills pad channels.
// ---------------------------------------------------------------------------
__global__ void gn_norm_relu_kernel(_Float16* __restrict__ x, const float* __restrict__ stats,
                                    const float* __restrict__ gw, const float* __restrict__ gb,
                                    int B, int HW, int CPAD, int C, int groups, int doRelu) {
    size_t total = (size_t)B * HW * CPAD;
    size_t i = (size_t)blockIdx.x * blockDim.x + threadIdx.x;
    if (i >= total) return;
    int c = (int)(i % CPAD);
    if (c >= C) { x[i] = (_Float16)0.f; return; }
    int b = (int)(i / ((size_t)HW * CPAD));
    int cg = C / groups;
    int g = c / cg;
    float N = (float)HW * (float)cg;
    float sum = stats[(b * groups + g) * 2 + 0];
    float sq = stats[(b * groups + g) * 2 + 1];
    float mean = sum / N;
    float var = sq / N - mean * mean;
    float inv = rsqrtf(var + 1e-5f);
    float v = ((float)x[i] - mean) * inv * gw[c] + gb[c];
    if (doRelu) v = fmaxf(v, 0.f);
    x[i] = (_Float16)v;
}

// ---------------------------------------------------------------------------
// Elementwise residual add (fp16)
// ---------------------------------------------------------------------------
__global__ void add_inplace_kernel(_Float16* __restrict__ a, const _Float16* __restrict__ b,
                                   size_t n) {
    size_t i = (size_t)blockIdx.x * blockDim.x + threadIdx.x;
    if (i < n) a[i] = (_Float16)((float)a[i] + (float)b[i]);
}

// ---------------------------------------------------------------------------
// conv5: 8 -> 2 channels, 3x3, pad 1 (tiny: scalar VALU). Output f32 planar.
// ---------------------------------------------------------------------------
__global__ void conv5_kernel(const _Float16* __restrict__ hin,    // NHWC, stride 16
                             const float* __restrict__ w5, const float* __restrict__ b5,
                             float* __restrict__ y5, int B, int H, int W) {
    size_t HW = (size_t)H * W;
    size_t idx = (size_t)blockIdx.x * blockDim.x + threadIdx.x;
    if (idx >= (size_t)B * HW) return;
    int b = (int)(idx / HW);
    int rem = (int)(idx % HW);
    int i = rem / W, j = rem % W;
    float acc0 = b5[0], acc1 = b5[1];
#pragma unroll
    for (int ky = 0; ky < 3; ++ky) {
        int iy = i + ky - 1;
        if ((unsigned)iy >= (unsigned)H) continue;
#pragma unroll
        for (int kx = 0; kx < 3; ++kx) {
            int ix = j + kx - 1;
            if ((unsigned)ix >= (unsigned)W) continue;
            const _Float16* p = hin + (((size_t)b * H + iy) * W + ix) * 16;
#pragma unroll
            for (int c = 0; c < 8; ++c) {
                float v = (float)p[c];
                acc0 += v * w5[((0 * 8 + c) * 3 + ky) * 3 + kx];
                acc1 += v * w5[((1 * 8 + c) * 3 + ky) * 3 + kx];
            }
        }
    }
    y5[((size_t)b * 2 + 0) * HW + (size_t)i * W + j] = acc0;
    y5[((size_t)b * 2 + 1) * HW + (size_t)i * W + j] = acc1;
}

// ---------------------------------------------------------------------------
// Joint bilateral blur (5x5, sigma_color=0.1, sigma_space=1.5, reflect pad,
// guidance = 2x tiled original => color dist = (2|d|)^2) + tanh + scale.
// ---------------------------------------------------------------------------
__global__ void bilateral_kernel(const float* __restrict__ y5, const float* __restrict__ x,
                                 float* __restrict__ out, int B, int H, int W) {
    size_t HW = (size_t)H * W;
    size_t idx = (size_t)blockIdx.x * blockDim.x + threadIdx.x;
    if (idx >= (size_t)B * HW) return;
    int b = (int)(idx / HW);
    int rem = (int)(idx % HW);
    int i = rem / W, j = rem % W;

    float g5[5]; float gs = 0.f;
#pragma unroll
    for (int k = 0; k < 5; ++k) { float t = (float)(k - 2); g5[k] = expf(-t * t / 4.5f); gs += g5[k]; }
#pragma unroll
    for (int k = 0; k < 5; ++k) g5[k] /= gs;

    const float* xb = x + (size_t)b * HW;
    const float* c0 = y5 + ((size_t)b * 2 + 0) * HW;
    const float* c1 = y5 + ((size_t)b * 2 + 1) * HW;
    float gc = xb[(size_t)i * W + j];
    float num0 = 0.f, num1 = 0.f, den = 0.f;
#pragma unroll
    for (int dy = 0; dy < 5; ++dy) {
        int iy = reflect_idx(i + dy - 2, H);
#pragma unroll
        for (int dx = 0; dx < 5; ++dx) {
            int ix = reflect_idx(j + dx - 2, W);
            size_t o = (size_t)iy * W + ix;
            float d = xb[o] - gc;
            float cd = 2.f * fabsf(d);                    // L1 over 2 identical channels
            float k = g5[dy] * g5[dx] * expf(-50.f * cd * cd);
            num0 += c0[o] * k;
            num1 += c1[o] * k;
            den += k;
        }
    }
    out[((size_t)b * 2 + 0) * HW + (size_t)i * W + j] = tanhf(num0 / den) * 0.436f;
    out[((size_t)b * 2 + 1) * HW + (size_t)i * W + j] = tanhf(num1 / den) * 0.615f;
}

// ---------------------------------------------------------------------------
// Host launcher
// ---------------------------------------------------------------------------
extern "C" void kernel_launch(void* const* d_in, const int* in_sizes, int n_in,
                              void* d_out, int out_size, void* d_ws, size_t ws_size,
                              hipStream_t stream) {
    (void)in_sizes; (void)n_in; (void)out_size; (void)ws_size;
    const int B = 4, H = 270, W = 512;
    const size_t HW = (size_t)H * W;
    const size_t BHW = (size_t)B * HW;

    const float* x   = (const float*)d_in[0];
    const float* w1  = (const float*)d_in[1];  const float* b1  = (const float*)d_in[2];
    const float* g1w = (const float*)d_in[3];  const float* g1b = (const float*)d_in[4];
    const float* w2  = (const float*)d_in[5];  const float* b2  = (const float*)d_in[6];
    const float* g2w = (const float*)d_in[7];  const float* g2b = (const float*)d_in[8];
    const float* w3  = (const float*)d_in[9];  const float* b3  = (const float*)d_in[10];
    const float* g3w = (const float*)d_in[11]; const float* g3b = (const float*)d_in[12];
    const float* w4  = (const float*)d_in[13]; const float* b4  = (const float*)d_in[14];
    const float* g4w = (const float*)d_in[15]; const float* g4b = (const float*)d_in[16];
    const float* w5  = (const float*)d_in[17]; const float* b5  = (const float*)d_in[18];
    const float* sw1 = (const float*)d_in[19]; const float* sb1 = (const float*)d_in[20];
    const float* sw2 = (const float*)d_in[21]; const float* sb2 = (const float*)d_in[22];

    char* ws = (char*)d_ws;
    size_t off = 0;
    auto alloc = [&](size_t bytes) -> char* {
        off = (off + 255) & ~(size_t)255;
        char* p = ws + off;
        off += bytes;
        return p;
    };

    float*     stats = (float*)alloc(256 * sizeof(float));
    float*     zpageF= (float*)alloc(512);                       // zeroed halo page
    const _Float16* zpage = (const _Float16*)zpageF;
    _Float16*  s     = (_Float16*)alloc(BHW * 128 * 2);
    // R1 region: SIFT scratch (ang+pooled) lives here, then reused for h1.
    char*      R1    = alloc(BHW * 64 * 2);
    float*     ang   = (float*)R1;                               // B*8*H*W f32
    float*     pooled= (float*)(R1 + ((BHW * 8 * 4 + 255) & ~(size_t)255));
    _Float16*  h1    = (_Float16*)R1;                            // B,H,W,64
    _Float16*  h2    = (_Float16*)alloc(BHW * 32 * 2);
    _Float16*  sk1   = (_Float16*)alloc(BHW * 32 * 2);
    _Float16*  h3    = (_Float16*)alloc(BHW * 32 * 2);           // 16 real + 16 pad
    _Float16*  h4    = (_Float16*)alloc(BHW * 16 * 2);           // 8 real + 8 pad
    _Float16*  sk2   = (_Float16*)alloc(BHW * 16 * 2);
    float*     y5    = (float*)alloc(BHW * 2 * 4);
    _Float16*  wp1   = (_Float16*)alloc((size_t)25 * 4 * 4 * 512 * 2);
    _Float16*  wp2   = (_Float16*)alloc((size_t)9 * 2 * 2 * 512 * 2);
    _Float16*  wp3   = (_Float16*)alloc((size_t)1 * 4 * 2 * 512 * 2);
    _Float16*  wp4   = (_Float16*)alloc((size_t)9 * 1 * 1 * 512 * 2);
    _Float16*  wp5   = (_Float16*)alloc((size_t)9 * 1 * 1 * 512 * 2);
    _Float16*  wp6   = (_Float16*)alloc((size_t)1 * 1 * 1 * 512 * 2);

    auto blocks = [](size_t n) { return (unsigned)((n + 255) / 256); };

    init_kernel<<<1, 256, 0, stream>>>(stats, zpageF);

    // --- dense SIFT -> s (NHWC fp16, C=128) ---
    sift_ang_kernel<<<blocks(BHW), 256, 0, stream>>>(x, ang, B, H, W);
    size_t ptot = (size_t)B * 8 * (H + 1) * (W + 1);
    sift_pool_kernel<<<blocks(ptot), 256, 0, stream>>>(ang, pooled, B, H, W);
    sift_desc_kernel<<<(unsigned)(BHW / 4), 128, 0, stream>>>(pooled, s, B, H, W);

    // --- weight packing into WMMA B-operand layout ---
    auto packL = [&](const float* w, _Float16* wp, int COUT, int CIN, int KS, int CINP, int NT) {
        size_t tot = (size_t)KS * KS * (CINP / 32) * NT * 512;
        pack_weights_kernel<<<blocks(tot), 256, 0, stream>>>(w, wp, COUT, CIN, KS, CINP, NT);
    };
    packL(w1,  wp1, 64, 128, 5, 128, 4);
    packL(w2,  wp2, 32, 64,  3, 64,  2);
    packL(sw1, wp3, 32, 128, 1, 128, 2);
    packL(w3,  wp4, 16, 32,  3, 32,  1);
    packL(w4,  wp5, 8,  16,  3, 32,  1);
    packL(sw2, wp6, 8,  32,  1, 32,  1);

    const unsigned MT2 = (unsigned)(B * H * (W / 32));  // 17280 supertiles (32 px)

    // --- conv1: 128 -> 64, 5x5 (dominant: ~226 GFLOP on WMMA) ---
    wmma_conv_kernel<128, 5, 2, 4, 64><<<MT2, 128, 0, stream>>>(s, wp1, b1, h1, zpage, B, H, W, 64);
    gn_stats_kernel<<<B * 8 * 64, 256, 0, stream>>>(h1, stats + 0, (int)HW, 64, 64, 8, 64);
    gn_norm_relu_kernel<<<blocks(BHW * 64), 256, 0, stream>>>(h1, stats + 0, g1w, g1b,
                                                              B, (int)HW, 64, 64, 8, 1);
    // --- conv2: 64 -> 32, 3x3 ---
    wmma_conv_kernel<64, 3, 1, 2, 32><<<MT2, 64, 0, stream>>>(h1, wp2, b2, h2, zpage, B, H, W, 32);
    // --- skip1: 128 -> 32, 1x1 ---
    wmma_conv_kernel<128, 1, 0, 2, 32><<<MT2, 64, 0, stream>>>(s, wp3, sb1, sk1, zpage, B, H, W, 32);
    gn_stats_kernel<<<B * 8 * 64, 256, 0, stream>>>(h2, stats + 64, (int)HW, 32, 32, 8, 64);
    gn_norm_relu_kernel<<<blocks(BHW * 32), 256, 0, stream>>>(h2, stats + 64, g2w, g2b,
                                                              B, (int)HW, 32, 32, 8, 1);
    add_inplace_kernel<<<blocks(BHW * 32), 256, 0, stream>>>(h2, sk1, BHW * 32);

    // --- conv3: 32 -> 16, 3x3 (output stored CPAD=32 for conv4's K=32) ---
    wmma_conv_kernel<32, 3, 1, 1, 16><<<MT2, 32, 0, stream>>>(h2, wp4, b3, h3, zpage, B, H, W, 32);
    gn_stats_kernel<<<B * 4 * 64, 256, 0, stream>>>(h3, stats + 128, (int)HW, 32, 16, 4, 64);
    gn_norm_relu_kernel<<<blocks(BHW * 32), 256, 0, stream>>>(h3, stats + 128, g3w, g3b,
                                                              B, (int)HW, 32, 16, 4, 1);
    // --- conv4: 16(pad32) -> 8, 3x3 ---
    wmma_conv_kernel<32, 3, 1, 1, 8><<<MT2, 32, 0, stream>>>(h3, wp5, b4, h4, zpage, B, H, W, 16);
    gn_stats_kernel<<<B * 4 * 64, 256, 0, stream>>>(h4, stats + 160, (int)HW, 16, 8, 4, 64);
    gn_norm_relu_kernel<<<blocks(BHW * 16), 256, 0, stream>>>(h4, stats + 160, g4w, g4b,
                                                              B, (int)HW, 16, 8, 4, 1);
    // --- skip2: 32 -> 8, 1x1 (reads post-skip1 h2) ---
    wmma_conv_kernel<32, 1, 0, 1, 8><<<MT2, 32, 0, stream>>>(h2, wp6, sb2, sk2, zpage, B, H, W, 16);
    add_inplace_kernel<<<blocks(BHW * 16), 256, 0, stream>>>(h4, sk2, BHW * 16);

    // --- conv5 (tiny) + joint bilateral + tanh + scale ---
    conv5_kernel<<<blocks(BHW), 256, 0, stream>>>(h4, w5, b5, y5, B, H, W);
    bilateral_kernel<<<blocks(BHW), 256, 0, stream>>>(y5, x, (float*)d_out, B, H, W);
}
